// FPLayer_33354716020953
// MI455X (gfx1250) — matrix operand order repaired and checked by reading, not compiled, not run on hardware
//
#include <hip/hip_runtime.h>

// ---------------- problem constants ----------------
#define BB   8
#define N1   8192
#define N2   2048
#define C1   128
#define C2   256
#define K0IN 384     // C1 + C2
#define H0   256     // MLP[0]
#define H1   128     // MLP[1]
#define MROWS (BB * N1)   // 65536
#define BN_EPS 1e-5f

typedef __attribute__((ext_vector_type(16))) __bf16 v16bf;
typedef __attribute__((ext_vector_type(8)))  __bf16 v8bf;
typedef __attribute__((ext_vector_type(8)))  float  v8f;
typedef __attribute__((ext_vector_type(4)))  unsigned int v4u;
typedef __attribute__((ext_vector_type(8)))  unsigned int v8u;

__device__ inline unsigned short f2bf(float f) {
    __bf16 h = (__bf16)f;
    return __builtin_bit_cast(unsigned short, h);
}

__device__ inline void store4bf(unsigned short* p, float a, float b, float c, float d) {
    union { unsigned short h[4]; uint2 u; } u;
    u.h[0] = f2bf(a); u.h[1] = f2bf(b); u.h[2] = f2bf(c); u.h[3] = f2bf(d);
    *(uint2*)p = u.u;
}

// ---- Tensor Data Mover: 2D tile (bf16) global -> LDS with HW padding ----
// pad_code_i: row length = 2^(code+1) DWORDs between pads
// pad_code_a: pad amount  = (code+1) DWORDs
__device__ inline void tdm_load_2d_bf16(unsigned lds_addr, const void* gptr,
                                        unsigned tile_w, unsigned tile_h,
                                        unsigned stride_elems,
                                        unsigned pad_code_i, unsigned pad_code_a) {
    unsigned long long ga = (unsigned long long)(size_t)gptr;
    v4u g0;
    g0[0] = 1u;                                              // count=1, user D#
    g0[1] = lds_addr;                                        // LDS byte address
    g0[2] = (unsigned)ga;                                    // global_addr[31:0]
    g0[3] = (unsigned)((ga >> 32) & 0x01FFFFFFu)             // global_addr[56:32]
          | 0x80000000u;                                     // type = 2 ("image")
    v8u g1;
    g1[0] = (1u << 16)                                       // data_size = 2 bytes
          | (1u << 20)                                       // pad_enable
          | (pad_code_i << 22)                               // pad_interval
          | (pad_code_a << 25);                              // pad_amount
    g1[1] = (tile_w & 0xFFFFu) << 16;                        // tensor_dim0[15:0]
    g1[2] = (tile_h & 0xFFFFu) << 16;                        // tensor_dim1[15:0]
    g1[3] = (tile_w & 0xFFFFu) << 16;                        // tile_dim0
    g1[4] = (tile_h & 0xFFFFu);                              // tile_dim1 (tile_dim2=0)
    g1[5] = stride_elems;                                    // tensor_dim0_stride[31:0]
    g1[6] = 0u;
    g1[7] = 0u;
    asm volatile("tensor_load_to_lds %0, %1" :: "s"(g0), "s"(g1) : "memory");
}

// ---------------- tiny utility kernels ----------------
__global__ __launch_bounds__(256) void zero_f32(float* p, int n) {
    int i = blockIdx.x * 256 + threadIdx.x;
    if (i < n) p[i] = 0.0f;
}

// dst[k*OUT + o] = bf16( W[o*IN + k] )   (row-major [K][N] B-matrix layout)
__global__ __launch_bounds__(256) void convert_weight(const float* __restrict__ W,
                                                      unsigned short* __restrict__ Wb,
                                                      int OUT, int IN) {
    int i = blockIdx.x * 256 + threadIdx.x;
    if (i >= OUT * IN) return;
    int k = i / OUT, o = i % OUT;
    Wb[i] = f2bf(W[o * IN + k]);
}

// ---------------- KNN + inverse-distance interp + concat -> bf16 ----------------
__global__ __launch_bounds__(256)
void knn_interp_concat(const float* __restrict__ xyz1, const float* __restrict__ xyz2,
                       const float* __restrict__ feat1, const float* __restrict__ feat2,
                       unsigned short* __restrict__ Xb) {
    __shared__ float sx[N2], sy[N2], sz[N2];
    const int b = blockIdx.y;
    const int n = blockIdx.x * 256 + threadIdx.x;

    const float* p2 = xyz2 + (size_t)b * N2 * 3;
    for (int j = threadIdx.x; j < N2; j += 256) {
        sx[j] = p2[j * 3 + 0];
        sy[j] = p2[j * 3 + 1];
        sz[j] = p2[j * 3 + 2];
    }
    __syncthreads();

    const size_t row = (size_t)b * N1 + n;
    const float px = xyz1[row * 3 + 0];
    const float py = xyz1[row * 3 + 1];
    const float pz = xyz1[row * 3 + 2];

    float d0 = 3.4e38f, d1 = 3.4e38f, d2v = 3.4e38f;
    int   i0 = 0, i1 = 0, i2 = 0;
    for (int j = 0; j < N2; ++j) {
        float dx = px - sx[j], dy = py - sy[j], dz = pz - sz[j];
        float d = dx * dx + dy * dy + dz * dz;
        if (d < d0)      { d2v = d1; i2 = i1; d1 = d0; i1 = i0; d0 = d; i0 = j; }
        else if (d < d1) { d2v = d1; i2 = i1; d1 = d;  i1 = j; }
        else if (d < d2v){ d2v = d;  i2 = j; }
    }
    float r0 = 1.0f / (sqrtf(fmaxf(d0,  1e-12f)) + 1e-8f);
    float r1 = 1.0f / (sqrtf(fmaxf(d1,  1e-12f)) + 1e-8f);
    float r2 = 1.0f / (sqrtf(fmaxf(d2v, 1e-12f)) + 1e-8f);
    float inv = 1.0f / (r0 + r1 + r2);
    float w0 = r0 * inv, w1 = r1 * inv, w2 = r2 * inv;

    unsigned short* out = Xb + row * K0IN;

    // feat1 -> cols [0, 128)
    const float4* f1 = (const float4*)(feat1 + row * C1);
    for (int c = 0; c < C1 / 4; ++c) {
        float4 v = f1[c];
        store4bf(out + c * 4, v.x, v.y, v.z, v.w);
    }
    // interpolated feat2 -> cols [128, 384)
    const float* f2 = feat2 + (size_t)b * N2 * C2;
    const float4* g0 = (const float4*)(f2 + (size_t)i0 * C2);
    const float4* g1 = (const float4*)(f2 + (size_t)i1 * C2);
    const float4* g2 = (const float4*)(f2 + (size_t)i2 * C2);
    for (int c = 0; c < C2 / 4; ++c) {
        float4 a = g0[c], bq = g1[c], cq = g2[c];
        store4bf(out + C1 + c * 4,
                 w0 * a.x + w1 * bq.x + w2 * cq.x,
                 w0 * a.y + w1 * bq.y + w2 * cq.y,
                 w0 * a.z + w1 * bq.z + w2 * cq.z,
                 w0 * a.w + w1 * bq.w + w2 * cq.w);
    }
}

// ---------------- bf16 WMMA GEMM with TDM tile staging ----------------
//   Y[M,N] = X[M,K] * Wb[K,N] + bias
#define LDA 40    // halves per A LDS row: 32 + 8 pad  (TDM: 16 DW rows + 4 DW pad)
#define LDB 136   // halves per B LDS row: 128 + 8 pad (TDM: 64 DW rows + 4 DW pad)

__global__ __launch_bounds__(256)
void gemm_bf16_wmma(const unsigned short* __restrict__ Xb,
                    const unsigned short* __restrict__ Wb,
                    const float* __restrict__ bias,
                    float* __restrict__ Y,
                    int M, int N, int K) {
    __shared__ unsigned short As[128 * LDA];
    __shared__ unsigned short Bs[32 * LDB];

    const int t    = threadIdx.x;
    const int m0   = blockIdx.x * 128;
    const int n0   = blockIdx.y * 128;
    const int wave = t >> 5;
    const int lane = t & 31;
    const int wm   = wave >> 2;        // 0..1 : 64-row slab
    const int wn   = wave & 3;         // 0..3 : 32-col slab
    const int lrow = lane & 15;
    const int khalf = lane >> 4;

    const unsigned ldsA = (unsigned)(size_t)(void*)As;   // low 32 bits = LDS offset
    const unsigned ldsB = (unsigned)(size_t)(void*)Bs;

    const v8f vzero = {0.f, 0.f, 0.f, 0.f, 0.f, 0.f, 0.f, 0.f};
    v8f acc[4][2];
    for (int i = 0; i < 4; ++i)
        for (int j = 0; j < 2; ++j)
            acc[i][j] = vzero;

    for (int k0 = 0; k0 < K; k0 += 32) {
        __syncthreads();                       // previous chunk's compute done
        if (wave == 0) {
            // A tile: 128 rows x 32 bf16, row stride K elems; pad 16DW rows by 4DW
            tdm_load_2d_bf16(ldsA, Xb + (size_t)m0 * K + k0, 32, 128, (unsigned)K, 3, 3);
            // B tile: 32 rows x 128 bf16, row stride N elems; pad 64DW rows by 4DW
            tdm_load_2d_bf16(ldsB, Wb + (size_t)k0 * N + n0, 128, 32, (unsigned)N, 5, 3);
            __builtin_amdgcn_s_wait_tensorcnt(0);
        }
        __syncthreads();                       // tiles visible to all waves

        // B fragments: lane -> N column (lane & 15); lanes 0-15 -> K 0-7 & 16-23, lanes 16-31 -> K 8-15 & 24-31
        v16bf bfrag[2];
        for (int tj = 0; tj < 2; ++tj) {
            const unsigned short* bp = Bs + (khalf * 8) * LDB + wn * 32 + tj * 16 + lrow;
            v16bf f;
            for (int i = 0; i < 8; ++i) { f[i] = __builtin_bit_cast(__bf16, bp[i * LDB]); f[8 + i] = __builtin_bit_cast(__bf16, bp[(16 + i) * LDB]); }
            bfrag[tj] = f;
        }
        for (int ti = 0; ti < 4; ++ti) {
            // A fragment: lanes 0-15 -> K 0-7 & 16-23, lanes 16-31 -> K 8-15 & 24-31
            const unsigned short* ap = As + (wm * 64 + ti * 16 + lrow) * LDA + khalf * 8;
            v8bf lo = *(const v8bf*)ap;
            v8bf hi = *(const v8bf*)(ap + 16);
            v16bf afrag;
            for (int i = 0; i < 8; ++i) { afrag[i] = lo[i]; afrag[8 + i] = hi[i]; }
            for (int tj = 0; tj < 2; ++tj) {
                acc[ti][tj] = __builtin_amdgcn_wmma_f32_16x16x32_bf16(
                    false, afrag, false, bfrag[tj], (short)0, acc[ti][tj], false, false);
            }
        }
    }

    // C layout: VGPR r -> M = r + 8*khalf ; N = lane & 15
    for (int ti = 0; ti < 4; ++ti) {
        for (int tj = 0; tj < 2; ++tj) {
            const int col = n0 + wn * 32 + tj * 16 + lrow;
            const float bv = bias[col];
            for (int r = 0; r < 8; ++r) {
                const int rowg = m0 + wm * 64 + ti * 16 + khalf * 8 + r;
                Y[(size_t)rowg * N + col] = acc[ti][tj][r] + bv;
            }
        }
    }
}

// ---------------- BN statistics (sum, sumsq per channel) ----------------
__global__ __launch_bounds__(256)
void bn_stats(const float* __restrict__ Y, float* __restrict__ stats,
              int M, int N, int rowsPerBlock) {
    const int tpc = 256 / N;                 // threads per channel
    const int ch  = threadIdx.x % N;
    const int sub = threadIdx.x / N;
    const int r0  = blockIdx.x * rowsPerBlock;
    float s = 0.f, ss = 0.f;
    for (int r = r0 + sub; r < r0 + rowsPerBlock; r += tpc) {
        float v = Y[(size_t)r * N + ch];
        s += v; ss += v * v;
    }
    unsafeAtomicAdd(&stats[ch], s);          // global_atomic_add_f32
    unsafeAtomicAdd(&stats[N + ch], ss);
}

// ---------------- BN + ReLU, bf16 output (feeds next GEMM) ----------------
__global__ __launch_bounds__(256)
void bn_relu_bf16(const float* __restrict__ Y, const float* __restrict__ stats,
                  const float* __restrict__ gamma, const float* __restrict__ beta,
                  unsigned short* __restrict__ out, int M, int N) {
    size_t i4 = (size_t)blockIdx.x * 256 + threadIdx.x;
    size_t i  = i4 * 4;
    if (i >= (size_t)M * N) return;
    const int ch = (int)(i % N);
    const float invM = 1.0f / (float)M;
    float4 v = *(const float4*)(Y + i);
    float vv[4] = {v.x, v.y, v.z, v.w};
    float o[4];
    for (int c = 0; c < 4; ++c) {
        int c2 = ch + c;
        float mean = stats[c2] * invM;
        float var  = stats[N + c2] * invM - mean * mean;
        float sc   = gamma[c2] * rsqrtf(var + BN_EPS);
        o[c] = fmaxf((vv[c] - mean) * sc + beta[c2], 0.0f);
    }
    store4bf(out + i, o[0], o[1], o[2], o[3]);
}

// ---------------- BN + ReLU, f32 output (final) ----------------
__global__ __launch_bounds__(256)
void bn_relu_f32(const float* __restrict__ Y, const float* __restrict__ stats,
                 const float* __restrict__ gamma, const float* __restrict__ beta,
                 float* __restrict__ out, int M, int N) {
    size_t i4 = (size_t)blockIdx.x * 256 + threadIdx.x;
    size_t i  = i4 * 4;
    if (i >= (size_t)M * N) return;
    const int ch = (int)(i % N);
    const float invM = 1.0f / (float)M;
    float4 v = *(const float4*)(Y + i);
    float vv[4] = {v.x, v.y, v.z, v.w};
    float4 o;
    float* op = &o.x;
    for (int c = 0; c < 4; ++c) {
        int c2 = ch + c;
        float mean = stats[c2] * invM;
        float var  = stats[N + c2] * invM - mean * mean;
        float sc   = gamma[c2] * rsqrtf(var + BN_EPS);
        op[c] = fmaxf((vv[c] - mean) * sc + beta[c2], 0.0f);
    }
    *(float4*)(out + i) = o;
}

// ---------------- host launch ----------------
extern "C" void kernel_launch(void* const* d_in, const int* in_sizes, int n_in,
                              void* d_out, int out_size, void* d_ws, size_t ws_size,
                              hipStream_t stream) {
    const float* xyz1   = (const float*)d_in[0];
    const float* xyz2   = (const float*)d_in[1];
    const float* feat1  = (const float*)d_in[2];
    const float* feat2  = (const float*)d_in[3];
    const float* W0     = (const float*)d_in[4];
    const float* b0     = (const float*)d_in[5];
    const float* gamma0 = (const float*)d_in[6];
    const float* beta0  = (const float*)d_in[7];
    const float* W1     = (const float*)d_in[8];
    const float* b1     = (const float*)d_in[9];
    const float* gamma1 = (const float*)d_in[10];
    const float* beta1  = (const float*)d_in[11];

    char* ws = (char*)d_ws;
    size_t off = 0;
    auto take = [&](size_t bytes) {
        size_t o = off;
        off += (bytes + 255) & ~(size_t)255;
        return o;
    };
    unsigned short* Xb    = (unsigned short*)(ws + take((size_t)MROWS * K0IN * 2));
    unsigned short* W0b   = (unsigned short*)(ws + take((size_t)K0IN * H0 * 2));
    unsigned short* W1b   = (unsigned short*)(ws + take((size_t)H0 * H1 * 2));
    float*          Y0    = (float*)(ws + take((size_t)MROWS * H0 * 4));
    unsigned short* X1b   = (unsigned short*)(ws + take((size_t)MROWS * H0 * 2));
    float*          Y1    = (float*)(ws + take((size_t)MROWS * H1 * 4));
    float*          stats = (float*)(ws + take((2 * H0 + 2 * H1) * 4));
    float* stats0 = stats;            // [sum(256) | sumsq(256)]
    float* stats1 = stats + 2 * H0;   // [sum(128) | sumsq(128)]

    // 1) zero BN stat accumulators (ws is not re-zeroed between replays)
    zero_f32<<<dim3((2 * H0 + 2 * H1 + 255) / 256), dim3(256), 0, stream>>>(
        stats, 2 * H0 + 2 * H1);

    // 2) weights -> bf16 B-layout [K][N]
    convert_weight<<<dim3((K0IN * H0 + 255) / 256), dim3(256), 0, stream>>>(W0, W0b, H0, K0IN);
    convert_weight<<<dim3((H0 * H1 + 255) / 256), dim3(256), 0, stream>>>(W1, W1b, H1, H0);

    // 3) KNN + interpolation + concat -> Xb bf16 [65536 x 384]
    knn_interp_concat<<<dim3(N1 / 256, BB), dim3(256), 0, stream>>>(
        xyz1, xyz2, feat1, feat2, Xb);

    // 4) layer 0 GEMM (TDM-staged WMMA bf16 -> f32): Y0 = Xb * W0b + b0
    gemm_bf16_wmma<<<dim3(MROWS / 128, H0 / 128), dim3(256), 0, stream>>>(
        Xb, W0b, b0, Y0, MROWS, H0, K0IN);

    // 5) BN stats + BN/ReLU -> bf16 activations
    bn_stats<<<dim3(MROWS / 256), dim3(256), 0, stream>>>(Y0, stats0, MROWS, H0, 256);
    bn_relu_bf16<<<dim3((unsigned)((size_t)MROWS * H0 / 4 / 256)), dim3(256), 0, stream>>>(
        Y0, stats0, gamma0, beta0, X1b, MROWS, H0);

    // 6) layer 1 GEMM: Y1 = X1b * W1b + b1
    gemm_bf16_wmma<<<dim3(MROWS / 128, H1 / 128), dim3(256), 0, stream>>>(
        X1b, W1b, b1, Y1, MROWS, H1, H0);

    // 7) BN stats + BN/ReLU -> final f32 output
    bn_stats<<<dim3(MROWS / 256), dim3(256), 0, stream>>>(Y1, stats1, MROWS, H1, 256);
    bn_relu_f32<<<dim3((unsigned)((size_t)MROWS * H1 / 4 / 256)), dim3(256), 0, stream>>>(
        Y1, stats1, gamma1, beta1, (float*)d_out, MROWS, H1);
}
